// GCN_2817498546214
// MI455X (gfx1250) — matrix operand order repaired and checked by reading, not compiled, not run on hardware
//
#include <hip/hip_runtime.h>
#include <hip/hip_bf16.h>

// ---------------------------------------------------------------------------
// GCN on gfx1250: WMMA bf16 GEMMs (f32 accumulate) + L2-resident atomic SpMM.
// Layer structure (reference):
//   t1 = X @ W1^T + b1 ; s1 = spmm(t1) ; h1 = relu(s1)
//   t2 = h1@ W2^T + b2 ; s2 = spmm(t2) ; h2 = relu(s2)
//   t3 = h2@ W3^T + b3 ; out = spmm(t3)
// ReLU is fused into the next GEMM's A-matrix load.
// ---------------------------------------------------------------------------

typedef __attribute__((ext_vector_type(16))) __bf16 v16bf;
typedef __attribute__((ext_vector_type(8)))  float  v8f;

static constexpr int CIN  = 256;
static constexpr int CHID = 256;
static constexpr int COUT = 64;

// --------------------------- weight fp32 -> bf16 ---------------------------
__global__ void cvt_bf16_kernel(const float* __restrict__ src,
                                __bf16* __restrict__ dst, int n) {
    int i = blockIdx.x * blockDim.x + threadIdx.x;
    if (i < n) dst[i] = (__bf16)src[i];
}

// ------------------------------ zero fill ----------------------------------
__global__ void fill_zero_kernel(float* __restrict__ p, size_t n4) {
    size_t i      = (size_t)blockIdx.x * blockDim.x + threadIdx.x;
    size_t stride = (size_t)gridDim.x * blockDim.x;
    float4 z = make_float4(0.f, 0.f, 0.f, 0.f);
    for (; i < n4; i += stride) ((float4*)p)[i] = z;
}

// ---------------------- WMMA GEMM: Y = A @ W^T + b -------------------------
// A: [N x K] fp32 row-major (optionally ReLU'd on load)
// W: [C x K] bf16 row-major (so B = W^T is K x C, loaded per-column = W row)
// One wave computes one 16x16 tile of Y. Requires N % 16 == 0, K % 32 == 0.
//
// A-fragment k order used here (16-bit A, 16x32): lane m = lane&15, half = lane>>4;
//   element e holds K = k0 + half*16 + e             (contiguous 16 floats)
// B-fragment k order used here (16-bit B, 32x16): lane n = lane&15, half = lane>>4;
//   element e holds K = k0 + half*16 + e             (contiguous 16 bf16)
// ISA D layout (f32 16x16): VGPR r holds row M = half*8 + r, col N = lane&15.
template<int RELU_IN, int K>
__global__ __launch_bounds__(256) void gemm_bias_kernel(
    const float* __restrict__ A, const __bf16* __restrict__ W,
    const float* __restrict__ bias, float* __restrict__ Y, int C) {
    const int lane    = threadIdx.x & 31;
    const int wave    = threadIdx.x >> 5;
    const int colTile = blockIdx.y * 8 + wave;
    if (colTile * 16 >= C) return;            // uniform per wave: safe w/ WMMA
    const int col0 = colTile * 16;
    const int m    = lane & 15;
    const int half = lane >> 4;

    const float*  arow = A + (size_t)(blockIdx.x * 16 + m) * K + half * 16;
    const __bf16* brow = W + (size_t)(col0 + m) * K + half * 16;

    auto cv = [](float x) -> __bf16 {
        if (RELU_IN) x = fmaxf(x, 0.0f);
        return (__bf16)x;
    };

    v8f acc = {};
#pragma unroll
    for (int k0 = 0; k0 < K; k0 += 32) {
        const float4 f0 = *(const float4*)(arow + k0);
        const float4 f1 = *(const float4*)(arow + k0 + 4);
        const float4 f2 = *(const float4*)(arow + k0 + 8);
        const float4 f3 = *(const float4*)(arow + k0 + 12);
        v16bf a;
        a[0]  = cv(f0.x); a[1]  = cv(f0.y); a[2]  = cv(f0.z); a[3]  = cv(f0.w);
        a[4]  = cv(f1.x); a[5]  = cv(f1.y); a[6]  = cv(f1.z); a[7]  = cv(f1.w);
        a[8]  = cv(f2.x); a[9]  = cv(f2.y); a[10] = cv(f2.z); a[11] = cv(f2.w);
        a[12] = cv(f3.x); a[13] = cv(f3.y); a[14] = cv(f3.z); a[15] = cv(f3.w);

        const v16bf b = *(const v16bf*)(brow + k0);   // 32B aligned vector load

        acc = __builtin_amdgcn_wmma_f32_16x16x32_bf16(
            /*neg_a=*/false, a, /*neg_b=*/false, b,
            /*c_mod=*/(short)0, acc, /*reuse_a=*/false, /*reuse_b=*/false);
    }

    const float bc = bias[col0 + m];
    float* yout = Y + (size_t)(blockIdx.x * 16 + half * 8) * C + col0 + m;
#pragma unroll
    for (int r = 0; r < 8; ++r)
        yout[(size_t)r * C] = acc[r] + bc;
}

// ------------------- SpMM scatter: Out[row] += val * H[col] ----------------
// One wave per edge; edge metadata is wave-uniform (scalarized by compiler).
// Gathers hit the 192MB L2 (activation matrix is 102.4MB); scatter uses
// no-return global_atomic_add_f32 resolved at L2 atomic units.
__global__ __launch_bounds__(256) void spmm_scatter_kernel(
    const float* __restrict__ H, const int* __restrict__ erow,
    const int* __restrict__ ecol, const float* __restrict__ eval,
    float* __restrict__ Out, int nEdges, int C) {
    const int  lane   = threadIdx.x & 31;
    const int  wpb    = blockDim.x >> 5;
    long       e      = (long)blockIdx.x * wpb + (threadIdx.x >> 5);
    const long stride = (long)gridDim.x * wpb;
    for (; e < nEdges; e += stride) {
        const int   r   = erow[e];
        const int   c   = ecol[e];
        const float v   = eval[e];
        const float* src = H   + (size_t)c * C;
        float*       dst = Out + (size_t)r * C;
        for (int j = lane * 4; j < C; j += 128) {
            const float4 x = *(const float4*)(src + j);
            atomicAdd(dst + j + 0, v * x.x);
            atomicAdd(dst + j + 1, v * x.y);
            atomicAdd(dst + j + 2, v * x.z);
            atomicAdd(dst + j + 3, v * x.w);
        }
    }
}

// ---------------------------------------------------------------------------
extern "C" void kernel_launch(void* const* d_in, const int* in_sizes, int n_in,
                              void* d_out, int out_size, void* d_ws, size_t ws_size,
                              hipStream_t stream) {
    const float* x    = (const float*)d_in[0];
    const int*   erow = (const int*)  d_in[1];
    const int*   ecol = (const int*)  d_in[2];
    const float* ev   = (const float*)d_in[3];
    const float* W1   = (const float*)d_in[4];
    const float* b1   = (const float*)d_in[5];
    const float* W2   = (const float*)d_in[6];
    const float* b2   = (const float*)d_in[7];
    const float* W3   = (const float*)d_in[8];
    const float* b3   = (const float*)d_in[9];

    const int N  = in_sizes[0] / CIN;   // 100000 (divisible by 16)
    const int nE = in_sizes[1];         // 3200000

    // Workspace layout: two ping-pong activation buffers + bf16 weights.
    char* ws = (char*)d_ws;
    const size_t bufBytes = (size_t)N * CHID * sizeof(float);   // 102.4 MB
    float*  bufA = (float*)ws;
    float*  bufB = (float*)(ws + bufBytes);
    __bf16* W1b  = (__bf16*)(ws + 2 * bufBytes);
    __bf16* W2b  = W1b + (size_t)CHID * CIN;
    __bf16* W3b  = W2b + (size_t)CHID * CHID;

    // 0) weights -> bf16
    cvt_bf16_kernel<<<dim3((CHID * CIN  + 255) / 256), 256, 0, stream>>>(W1, W1b, CHID * CIN);
    cvt_bf16_kernel<<<dim3((CHID * CHID + 255) / 256), 256, 0, stream>>>(W2, W2b, CHID * CHID);
    cvt_bf16_kernel<<<dim3((COUT * CHID + 255) / 256), 256, 0, stream>>>(W3, W3b, COUT * CHID);

    const dim3 gemmGridWide((N + 15) / 16, 2);   // 16 col tiles / 8 waves
    const dim3 gemmGridOut ((N + 15) / 16, 1);   // 4 col tiles (waves 4-7 idle)
    const dim3 spmmGrid((nE + 7) / 8);           // 8 waves per 256-thread block
    const size_t n4Hid = (size_t)N * CHID / 4;
    const size_t n4Out = (size_t)N * COUT / 4;
    const int fillBlocksHid = (int)((n4Hid + 255) / 256) < 16384 ? (int)((n4Hid + 255) / 256) : 16384;
    const int fillBlocksOut = (int)((n4Out + 255) / 256) < 16384 ? (int)((n4Out + 255) / 256) : 16384;

    // Layer 1: t1 = X @ W1^T + b1 ; s1 = spmm(t1)
    gemm_bias_kernel<0, 256><<<gemmGridWide, 256, 0, stream>>>(x, W1b, b1, bufA, CHID);
    fill_zero_kernel<<<fillBlocksHid, 256, 0, stream>>>(bufB, n4Hid);
    spmm_scatter_kernel<<<spmmGrid, 256, 0, stream>>>(bufA, erow, ecol, ev, bufB, nE, CHID);

    // Layer 2: t2 = relu(s1) @ W2^T + b2 ; s2 = spmm(t2)   (relu fused in A-load)
    gemm_bias_kernel<1, 256><<<gemmGridWide, 256, 0, stream>>>(bufB, W2b, b2, bufA, CHID);
    fill_zero_kernel<<<fillBlocksHid, 256, 0, stream>>>(bufB, n4Hid);
    spmm_scatter_kernel<<<spmmGrid, 256, 0, stream>>>(bufA, erow, ecol, ev, bufB, nE, CHID);

    // Layer 3: t3 = relu(s2) @ W3^T + b3 ; out = spmm(t3)
    gemm_bias_kernel<1, 256><<<gemmGridOut, 256, 0, stream>>>(bufB, W3b, b3, bufA, COUT);
    fill_zero_kernel<<<fillBlocksOut, 256, 0, stream>>>((float*)d_out, n4Out);
    spmm_scatter_kernel<<<spmmGrid, 256, 0, stream>>>(bufA, erow, ecol, ev, (float*)d_out, nE, COUT);
}
